// GCN_69123203662067
// MI455X (gfx1250) — matrix operand order repaired and checked by reading, not compiled, not run on hardware
//
#include <hip/hip_runtime.h>
#include <math.h>

typedef __bf16 bf16_t;
typedef __attribute__((ext_vector_type(16))) __bf16 v16bf;
typedef __attribute__((ext_vector_type(8)))  float  v8f;

#define WPB 8  // waves per block (256 threads, wave32)

__device__ __forceinline__ float selu_f(float x) {
    const float alpha = 1.6732632423543772848f;
    const float scale = 1.0507009873554804934f;
    return x > 0.0f ? scale * x : scale * alpha * (expf(x) - 1.0f);
}

// ---------------- prep / elementwise kernels ----------------

__global__ void k_fill(float* __restrict__ p, float v, long long n) {
    long long i = (long long)blockIdx.x * blockDim.x + threadIdx.x;
    if (i < n) p[i] = v;
}

__global__ void k_deg_accum(const int* __restrict__ ei, float* __restrict__ deg, int E) {
    int i = blockIdx.x * blockDim.x + threadIdx.x;
    if (i < E) atomicAdd(&deg[ei[E + i]], 1.0f);
}

__global__ void k_deg_to_dis(float* __restrict__ d, int n) {
    int i = blockIdx.x * blockDim.x + threadIdx.x;
    if (i < n) d[i] = rsqrtf(d[i]);   // deg >= 1 due to self loop
}

// Pack f32 weight [Ksrc,N] into WMMA B-fragment-major bf16 layout for Kpad:
// dst[((ct*ksteps + ks)*32 + lane)*16 + j] = W[ks*32 + (lane>>4)*16 + j][ct*16 + (lane&15)]
// (zero-padded for k >= Ksrc). Each lane's 16 bf16 land contiguous (32B).
__global__ void k_pack_w(const float* __restrict__ src, bf16_t* __restrict__ dst,
                         int Ksrc, int Kpad, int N) {
    int p = blockIdx.x * blockDim.x + threadIdx.x;
    int total = Kpad * N;
    if (p >= total) return;
    int j    = p & 15;
    int lane = (p >> 4) & 31;
    int t2   = p >> 9;
    int ksteps = Kpad >> 5;
    int ks = t2 % ksteps;
    int ct = t2 / ksteps;
    int k = (ks << 5) + ((lane >> 4) << 4) + j;
    int n = (ct << 4) + (lane & 15);
    float v = (k < Ksrc) ? src[(size_t)k * N + n] : 0.0f;
    dst[p] = (bf16_t)v;
}

// Zero-pad z [N,16] -> [N,32] so the decoder GEMM has K%32==0 (branch-free path).
__global__ void k_pad32(const float* __restrict__ src, float* __restrict__ dst, int Nn) {
    int i = blockIdx.x * blockDim.x + threadIdx.x;
    if (i >= Nn * 32) return;
    int row = i >> 5, c = i & 31;
    dst[i] = (c < 16) ? src[(size_t)row * 16 + c] : 0.0f;
}

// ---------------- WMMA GEMM: C = act(A @ W + bias) ----------------
// A: [M,K] f32 row-major, K % 32 == 0. Wp: packed fragment-major bf16 (k_pack_w).
// One wave computes 16 rows x (16*NT) cols. NCOLS compile-time -> immediate offsets.
template <int NT, int NCOLS>
__launch_bounds__(256)
__global__ void gemm_wmma(const float* __restrict__ A, const bf16_t* __restrict__ Wp,
                          const float* __restrict__ bias, float* __restrict__ C,
                          int M, int K, int act /*0 none, 1 relu*/) {
    const int lane = threadIdx.x & 31;
    const int wave = threadIdx.x >> 5;
    const int h    = lane >> 4;   // half-wave select
    const int ln   = lane & 15;   // A row / B-C-D column within tile
    constexpr int colgroups = NCOLS / (16 * NT);
    const int tile     = blockIdx.x * WPB + wave;
    const int row_tile = tile / colgroups;
    const int cg       = tile % colgroups;
    if (row_tile * 16 >= M) return;
    const int row0 = row_tile * 16;

    int ar = row0 + ln;
    if (ar >= M) ar = M - 1;                 // clamp (loads only; stores guarded)
    const float* arow = A + (size_t)ar * K;
    const int ksteps = K >> 5;

    const v16bf* wbase = reinterpret_cast<const v16bf*>(Wp);
    v8f acc[NT] = {};

    for (int kt = 0; kt < ksteps; ++kt) {
        // --- A fragment: one contiguous 64B run per lane (k = 16h + s), packed cvt to bf16 ---
        const float4* ap = (const float4*)(arow + (kt << 5) + (h << 4));
        float4 q0 = ap[0];
        float4 q1 = ap[1];
        float4 q2 = ap[2];   // +8 floats
        float4 q3 = ap[3];
        v16bf a;
        a[0]  = (bf16_t)q0.x; a[1]  = (bf16_t)q0.y; a[2]  = (bf16_t)q0.z; a[3]  = (bf16_t)q0.w;
        a[4]  = (bf16_t)q1.x; a[5]  = (bf16_t)q1.y; a[6]  = (bf16_t)q1.z; a[7]  = (bf16_t)q1.w;
        a[8]  = (bf16_t)q2.x; a[9]  = (bf16_t)q2.y; a[10] = (bf16_t)q2.z; a[11] = (bf16_t)q2.w;
        a[12] = (bf16_t)q3.x; a[13] = (bf16_t)q3.y; a[14] = (bf16_t)q3.z; a[15] = (bf16_t)q3.w;
#pragma unroll
        for (int t = 0; t < NT; ++t) {
            // --- B fragment: one 32B contiguous load per lane (packed layout) ---
            v16bf b = wbase[((size_t)(cg * NT + t) * ksteps + kt) * 32 + lane];
            acc[t] = __builtin_amdgcn_wmma_f32_16x16x32_bf16(
                false, a, false, b, (short)0, acc[t], false, false);
        }
    }

    // --- epilogue: bias + activation, doc C/D layout (VGPR r -> row h*8+r) ---
    const bool full = (row0 + 16 <= M);
#pragma unroll
    for (int t = 0; t < NT; ++t) {
        int col = cg * (16 * NT) + t * 16 + ln;
        float bv = bias ? bias[col] : 0.0f;
        float* crow = C + (size_t)(row0 + h * 8) * NCOLS + col;
        if (full) {
#pragma unroll
            for (int r = 0; r < 8; ++r) {
                float v = acc[t][r] + bv;
                if (act == 1) v = fmaxf(v, 0.0f);
                crow[(size_t)r * NCOLS] = v;
            }
        } else {
#pragma unroll
            for (int r = 0; r < 8; ++r) {
                if (row0 + h * 8 + r < M) {
                    float v = acc[t][r] + bv;
                    if (act == 1) v = fmaxf(v, 0.0f);
                    crow[(size_t)r * NCOLS] = v;
                }
            }
        }
    }
}

// ---------------- edge scatter: agg[dst] += hw[src] * dis[src]*dis[dst] ----------------
__global__ void k_scatter(const int* __restrict__ ei, const float* __restrict__ hw,
                          const float* __restrict__ dis, float* __restrict__ agg,
                          int E, int F) {
    long long tid = (long long)blockIdx.x * blockDim.x + threadIdx.x;
    const int chunks = F >> 2;
    long long total = (long long)E * chunks;
    if (tid >= total) return;
    int e  = (int)(tid / chunks);
    int fc = (int)(tid % chunks) << 2;
    int src = ei[e];
    int dst = ei[E + e];
    float w = dis[src] * dis[dst];
    const float4 v = *(const float4*)(hw + (size_t)src * F + fc);
    float* o = agg + (size_t)dst * F + fc;
    atomicAdd(o + 0, v.x * w);
    atomicAdd(o + 1, v.y * w);
    atomicAdd(o + 2, v.z * w);
    atomicAdd(o + 3, v.w * w);
}

// ---------------- conv epilogue: out = (1-s)*act(agg + hw*dis^2 + b) + s*blend ----------------
__global__ void k_conv_epi(const float* __restrict__ agg, const float* __restrict__ hw,
                           const float* __restrict__ dis, const float* __restrict__ bias,
                           const float* __restrict__ blend_in, float* __restrict__ out,
                           int Nn, int F, int use_selu, float sigma) {
    long long i = (long long)blockIdx.x * blockDim.x + threadIdx.x;
    long long total = (long long)Nn * F;
    if (i >= total) return;
    int row = (int)(i / F);
    int col = (int)(i % F);
    float d = dis[row];
    float v = agg[i] + hw[i] * d * d + bias[col];  // self-loop folded in
    if (use_selu) v = selu_f(v);
    out[i] = (1.0f - sigma) * v + sigma * blend_in[i];
}

// ---------------- soft assignment ----------------
__global__ void k_reduce_sum(const float* __restrict__ g, float* __restrict__ out, long long n) {
    __shared__ float sm[256];
    float acc = 0.0f;
    for (long long i = (long long)blockIdx.x * blockDim.x + threadIdx.x; i < n;
         i += (long long)gridDim.x * blockDim.x)
        acc += g[i];
    sm[threadIdx.x] = acc;
    __syncthreads();
    for (int s = 128; s > 0; s >>= 1) {
        if (threadIdx.x < s) sm[threadIdx.x] += sm[threadIdx.x + s];
        __syncthreads();
    }
    if (threadIdx.x == 0) atomicAdd(out, sm[0]);
}

__global__ void k_soft_assign(float* __restrict__ g, const float* __restrict__ sumptr, int Nn) {
    int i = blockIdx.x * blockDim.x + threadIdx.x;
    if (i >= Nn) return;
    float inv = 1.0f / (*sumptr);
    float t[16];
    float s2 = 0.0f;
#pragma unroll
    for (int f = 0; f < 16; ++f) {
        float v = tanhf(g[(size_t)i * 16 + f] * inv);
        v = v * v;
        t[f] = v;
        s2 += v * v;
    }
    float nrm = fmaxf(sqrtf(s2), 1e-12f);
#pragma unroll
    for (int f = 0; f < 16; ++f) g[(size_t)i * 16 + f] = t[f] / nrm;
}

// ---------------- host orchestration ----------------
extern "C" void kernel_launch(void* const* d_in, const int* in_sizes, int n_in,
                              void* d_out, int out_size, void* d_ws, size_t ws_size,
                              hipStream_t stream) {
    const int FIN = 512, H1 = 256, H2 = 128, FOUT = 16;
    const float SIGMA = 0.5f;

    const float* x  = (const float*)d_in[0];
    const int*   ei = (const int*)d_in[1];
    const int N = in_sizes[0] / FIN;
    const int E = in_sizes[1] / 2;
    const float* W1  = (const float*)d_in[2];  const float* b1  = (const float*)d_in[3];
    const float* W2  = (const float*)d_in[4];  const float* b2  = (const float*)d_in[5];
    const float* W3  = (const float*)d_in[6];  const float* b3  = (const float*)d_in[7];
    const float* E1w = (const float*)d_in[8];  const float* E1b = (const float*)d_in[9];
    const float* E2w = (const float*)d_in[10]; const float* E2b = (const float*)d_in[11];
    const float* E3w = (const float*)d_in[12]; const float* E3b = (const float*)d_in[13];
    const float* D1w = (const float*)d_in[14]; const float* D1b = (const float*)d_in[15];
    const float* D2w = (const float*)d_in[16]; const float* D2b = (const float*)d_in[17];
    const float* D3w = (const float*)d_in[18]; const float* D3b = (const float*)d_in[19];

    float* xr = (float*)d_out;                   // [N,512]
    float* z  = xr + (size_t)N * FIN;            // [N,16]
    float* g  = z  + (size_t)N * FOUT;           // [N,16]

    char* ws = (char*)d_ws;
    size_t off = 0;
    auto alloc = [&](size_t bytes) -> void* {
        off = (off + 255) & ~(size_t)255;
        void* p = ws + off;
        off += bytes;
        return p;
    };
    float*  dis   = (float*)alloc((size_t)N * 4);
    float*  sumS  = (float*)alloc(4);
    bf16_t* W1p   = (bf16_t*)alloc((size_t)FIN * H1 * 2);
    bf16_t* W2p   = (bf16_t*)alloc((size_t)H1 * H2 * 2);
    bf16_t* W3p   = (bf16_t*)alloc((size_t)H2 * FOUT * 2);
    bf16_t* E1p   = (bf16_t*)alloc((size_t)FIN * H1 * 2);
    bf16_t* E2p   = (bf16_t*)alloc((size_t)H1 * H2 * 2);
    bf16_t* E3p   = (bf16_t*)alloc((size_t)H2 * FOUT * 2);
    bf16_t* D1p   = (bf16_t*)alloc((size_t)32 * H2 * 2);   // K padded 16 -> 32
    bf16_t* D2p   = (bf16_t*)alloc((size_t)H2 * H1 * 2);
    bf16_t* D3p   = (bf16_t*)alloc((size_t)H1 * FIN * 2);
    float*  bufA  = (float*)alloc((size_t)N * H1 * 4);     // xe1 -> h1 -> xr2
    float*  bufB  = (float*)alloc((size_t)N * H1 * 4);     // hw1/hw2/hw3 -> xr1
    float*  bufC  = (float*)alloc((size_t)N * H1 * 4);     // agg
    float*  bufE  = (float*)alloc((size_t)N * H2 * 4);     // xe2 -> h2
    float*  bufZ  = (float*)alloc((size_t)N * 32 * 4);     // z zero-padded to K=32

    auto pack = [&](const float* s, bf16_t* d, int Ksrc, int Kpad, int Nc) {
        int n = Kpad * Nc;
        k_pack_w<<<(n + 255) / 256, 256, 0, stream>>>(s, d, Ksrc, Kpad, Nc);
    };
    auto zero = [&](float* p, long long n) {
        k_fill<<<(int)((n + 255) / 256), 256, 0, stream>>>(p, 0.0f, n);
    };
    auto blocks = [&](int M, int ncols, int nt) {
        long long tiles = (long long)((M + 15) / 16) * (ncols / (16 * nt));
        return (int)((tiles + WPB - 1) / WPB);
    };
    auto scatter = [&](const float* hw, float* agg, int F) {
        long long total = (long long)E * (F >> 2);
        k_scatter<<<(int)((total + 255) / 256), 256, 0, stream>>>(ei, hw, dis, agg, E, F);
    };
    auto epi = [&](const float* agg, const float* hw, const float* bias,
                   const float* blend_in, float* out, int F, int use_selu) {
        long long total = (long long)N * F;
        k_conv_epi<<<(int)((total + 255) / 256), 256, 0, stream>>>(
            agg, hw, dis, bias, blend_in, out, N, F, use_selu, SIGMA);
    };

    // weight packing (bf16 fragment-major for WMMA)
    pack(W1, W1p, FIN, FIN, H1);   pack(W2, W2p, H1, H1, H2);  pack(W3, W3p, H2, H2, FOUT);
    pack(E1w, E1p, FIN, FIN, H1);  pack(E2w, E2p, H1, H1, H2); pack(E3w, E3p, H2, H2, FOUT);
    pack(D1w, D1p, FOUT, 32, H2);  pack(D2w, D2p, H2, H2, H1); pack(D3w, D3p, H1, H1, FIN);

    // degree -> dis = rsqrt(deg), self-loop included via init=1
    k_fill<<<(N + 255) / 256, 256, 0, stream>>>(dis, 1.0f, N);
    k_deg_accum<<<(E + 255) / 256, 256, 0, stream>>>(ei, dis, E);
    k_deg_to_dis<<<(N + 255) / 256, 256, 0, stream>>>(dis, N);

    // encoder stage 1 + conv1
    gemm_wmma<4, 256><<<blocks(N, 256, 4), 256, 0, stream>>>(x, E1p, E1b, bufA, N, FIN, 1);    // xe1
    gemm_wmma<4, 256><<<blocks(N, 256, 4), 256, 0, stream>>>(x, W1p, nullptr, bufB, N, FIN, 0);// hw1
    gemm_wmma<4, 128><<<blocks(N, 128, 4), 256, 0, stream>>>(bufA, E2p, E2b, bufE, N, H1, 1);  // xe2
    zero(bufC, (long long)N * H1);
    scatter(bufB, bufC, H1);
    epi(bufC, bufB, b1, bufA, bufA, H1, 1);          // h1 = (1-s)*selu(conv1)+s*xe1 (in place)

    // conv2 + encoder stage 3
    gemm_wmma<4, 128><<<blocks(N, 128, 4), 256, 0, stream>>>(bufA, W2p, nullptr, bufB, N, H1, 0); // hw2
    gemm_wmma<1, 16><<<blocks(N, 16, 1), 256, 0, stream>>>(bufE, E3p, E3b, z, N, H2, 0);          // z
    zero(bufC, (long long)N * H2);
    scatter(bufB, bufC, H2);
    epi(bufC, bufB, b2, bufE, bufE, H2, 1);          // h2 (in place over xe2)

    // conv3 + blend with z -> raw g
    gemm_wmma<1, 16><<<blocks(N, 16, 1), 256, 0, stream>>>(bufE, W3p, nullptr, bufB, N, H2, 0);   // hw3
    zero(bufC, (long long)N * FOUT);
    scatter(bufB, bufC, FOUT);
    epi(bufC, bufB, b3, z, g, FOUT, 0);              // g = (1-s)*conv3 + s*z

    // decoder
    k_pad32<<<(N * 32 + 255) / 256, 256, 0, stream>>>(z, bufZ, N);
    gemm_wmma<4, 128><<<blocks(N, 128, 4), 256, 0, stream>>>(bufZ, D1p, D1b, bufB, N, 32, 1);     // xr1
    gemm_wmma<4, 256><<<blocks(N, 256, 4), 256, 0, stream>>>(bufB, D2p, D2b, bufA, N, H2, 1);     // xr2
    gemm_wmma<4, 512><<<blocks(N, 512, 4), 256, 0, stream>>>(bufA, D3p, D3b, xr, N, H1, 0);       // xr

    // soft assignment on g (in place)
    zero(sumS, 1);
    k_reduce_sum<<<1024, 256, 0, stream>>>(g, sumS, (long long)N * FOUT);
    k_soft_assign<<<(N + 255) / 256, 256, 0, stream>>>(g, sumS, N);
}
